// TransformerBlockRes_30021821399848
// MI455X (gfx1250) — matrix operand order repaired and checked by reading, not compiled, not run on hardware
//
#include <hip/hip_runtime.h>

// ---------------------------------------------------------------- types
typedef __attribute__((ext_vector_type(16))) __bf16 v16bf;
typedef __attribute__((ext_vector_type(8)))  float  v8f;

struct U4 { unsigned int x, y, z, w; };

__device__ inline unsigned short f2bf(float f) {
  union { float f; unsigned int u; } c; c.f = f;
  unsigned int u = c.u;
  unsigned int r = (u + 0x7FFFu + ((u >> 16) & 1u)) >> 16;  // RNE
  return (unsigned short)r;
}
__device__ inline float bf2f(unsigned short h) {
  union { unsigned int u; float f; } c; c.u = ((unsigned int)h) << 16;
  return c.f;
}
__device__ inline float elu1(float x) { return x > 0.f ? x : (__expf(x) - 1.f); }

// A fragment, 16-bit 16x32 (ISA 7.12.2): lanes 0-15 -> K[0..7] in v0-3, K[16..23] in v4-7;
// lanes 16-31 -> K[8..15], K[24..31].  row = contiguous K in memory.
__device__ inline v16bf fragA(const unsigned short* row, int half) {
  union { U4 q[2]; v16bf v; } f;
  f.q[0] = *reinterpret_cast<const U4*>(row + half * 8);
  f.q[1] = *reinterpret_cast<const U4*>(row + half * 8 + 16);
  return f.v;
}
// B fragment, 16-bit 32x16: lanes 0-15 -> K[0..7], K[16..23]; lanes 16-31 -> K[8..15], K[24..31].
// row = B^T row (contiguous K).
__device__ inline v16bf fragB(const unsigned short* row, int half) {
  union { U4 q[2]; v16bf v; } f;
  f.q[0] = *reinterpret_cast<const U4*>(row + half * 8);
  f.q[1] = *reinterpret_cast<const U4*>(row + half * 8 + 16);
  return f.v;
}

__device__ inline float wave_sum(float v) {
#pragma unroll
  for (int m = 16; m >= 1; m >>= 1) v += __shfl_xor(v, m, 32);
  return v;
}

#define NPTS 4096
#define CIN 128
#define HDIM 512
#define KNN 20
#define NNB 21      // K + self loop
#define GPB 3       // nodes (groups) per fused block
#define MROWS 64    // padded M panel (3*21 = 63 valid rows)

// ---------------------------------------------------------------- prep
__global__ void k_cast_bf16(const float* __restrict__ src, unsigned short* __restrict__ dst, int n) {
  int i = blockIdx.x * 256 + threadIdx.x;
  if (i < n) dst[i] = f2bf(src[i]);
}

// src [K,N] f32 row-major -> dst [N,K] bf16 row-major
__global__ void k_transpose_bf16(const float* __restrict__ src, unsigned short* __restrict__ dst,
                                 int K, int N) {
  int e = blockIdx.x * 256 + threadIdx.x;
  if (e < K * N) { int n = e / K, k = e - n * K; dst[e] = f2bf(src[k * N + n]); }
}

// WcatT [384][128] bf16: rows 0-127 = W_lin cols, 128-255 = W_src cols, 256-383 = W_dst cols
__global__ void k_wcat(const float* __restrict__ Wl, const float* __restrict__ Ws,
                       const float* __restrict__ Wd, unsigned short* __restrict__ dst) {
  int e = blockIdx.x * 256 + threadIdx.x;
  if (e < 384 * 128) {
    int n = e >> 7, k = e & 127;
    const float* W = (n < 128) ? Wl : (n < 256 ? Ws : Wd);
    dst[e] = f2bf(W[k * 128 + (n & 127)]);
  }
}

// ---------------------------------------------------------------- knn
__global__ void k_knn(const float* __restrict__ pos, int* __restrict__ idx) {
  __shared__ float tile[128 * 3];
  int i = blockIdx.x * 128 + threadIdx.x;
  float px = pos[i * 3 + 0], py = pos[i * 3 + 1], pz = pos[i * 3 + 2];
  float bd[KNN]; int bi[KNN];
#pragma unroll
  for (int s = 0; s < KNN; ++s) { bd[s] = 1e30f; bi[s] = i; }
  for (int t = 0; t < NPTS; t += 128) {
    __syncthreads();
    int j = t + threadIdx.x;
    tile[threadIdx.x * 3 + 0] = pos[j * 3 + 0];
    tile[threadIdx.x * 3 + 1] = pos[j * 3 + 1];
    tile[threadIdx.x * 3 + 2] = pos[j * 3 + 2];
    __syncthreads();
    for (int jj = 0; jj < 128; ++jj) {
      int j2 = t + jj;
      float dx = px - tile[jj * 3 + 0];
      float dy = py - tile[jj * 3 + 1];
      float dz = pz - tile[jj * 3 + 2];
      float d = dx * dx + dy * dy + dz * dz;
      if (j2 == i) d = 1e30f;
      if (d < bd[KNN - 1]) {
        float cd = d; int ci = j2;
#pragma unroll
        for (int s = 0; s < KNN; ++s) {  // swap-down insertion, statically indexed -> VGPRs
          if (cd < bd[s]) { float td = bd[s]; int ti = bi[s]; bd[s] = cd; bi[s] = ci; cd = td; ci = ti; }
        }
      }
    }
  }
#pragma unroll
  for (int s = 0; s < KNN; ++s) idx[i * NNB + s] = bi[s];
  idx[i * NNB + KNN] = i;  // self loop
}

// ---------------------------------------------------------------- node GEMM: vsd[4096,384] = x @ [Wl|Ws|Wd]
__global__ __launch_bounds__(256) void k_node_gemm(const unsigned short* __restrict__ xb,
                                                   const unsigned short* __restrict__ wT,
                                                   float* __restrict__ vsd) {
  int lane = threadIdx.x & 31, wid = threadIdx.x >> 5;
  int half = lane >> 4, lm = lane & 15;
  int gw = blockIdx.x * 8 + wid;          // 6144 waves = 256 mtiles * 24 ntiles
  int mt = gw / 24, nt = gw - mt * 24;
  const unsigned short* arow = xb + (mt * 16 + lm) * CIN;
  const unsigned short* brow = wT + (nt * 16 + lm) * CIN;
  v8f acc = {0.f, 0.f, 0.f, 0.f, 0.f, 0.f, 0.f, 0.f};
#pragma unroll
  for (int k = 0; k < CIN; k += 32) {
    v16bf a = fragA(arow + k, half);
    v16bf b = fragB(brow + k, half);
    acc = __builtin_amdgcn_wmma_f32_16x16x32_bf16(false, a, false, b, (short)0, acc, false, false);
  }
  int n = nt * 16 + lm;
#pragma unroll
  for (int r = 0; r < 8; ++r)
    vsd[(mt * 16 + half * 8 + r) * 384 + n] = acc[r];
}

// ---------------------------------------------------------------- LN helpers (wave-per-row)
__device__ inline void ln_elu_512_bf(unsigned short* buf, const float* __restrict__ g,
                                     const float* __restrict__ b, int lane, int wid, int nrows) {
  for (int r = wid; r < nrows; r += 8) {
    float s = 0.f, sq = 0.f;
#pragma unroll
    for (int c = lane; c < HDIM; c += 32) { float x = bf2f(buf[r * HDIM + c]); s += x; sq += x * x; }
    s = wave_sum(s); sq = wave_sum(sq);
    float mean = s * (1.f / HDIM);
    float inv = rsqrtf(sq * (1.f / HDIM) - mean * mean + 1e-5f);
#pragma unroll
    for (int c = lane; c < HDIM; c += 32) {
      float x = bf2f(buf[r * HDIM + c]);
      buf[r * HDIM + c] = f2bf(elu1((x - mean) * inv * g[c] + b[c]));
    }
  }
}
__device__ inline void ln_elu_128_f32(float* buf, const float* __restrict__ g,
                                      const float* __restrict__ b, int lane, int wid, int nrows) {
  for (int r = wid; r < nrows; r += 8) {
    float s = 0.f, sq = 0.f;
#pragma unroll
    for (int c = lane; c < CIN; c += 32) { float x = buf[r * CIN + c]; s += x; sq += x * x; }
    s = wave_sum(s); sq = wave_sum(sq);
    float mean = s * (1.f / CIN);
    float inv = rsqrtf(sq * (1.f / CIN) - mean * mean + 1e-5f);
#pragma unroll
    for (int c = lane; c < CIN; c += 32) {
      float x = buf[r * CIN + c];
      buf[r * CIN + c] = elu1((x - mean) * inv * g[c] + b[c]);
    }
  }
}

// ---------------------------------------------------------------- fused per-3-node edge pipeline
// LDS: s_act [64][512] bf16 (64KB) | s_delta [64][128] f32 (32KB) | s_alpha [64][128] f32 (32KB)
//      s_ain [64][128] bf16 (16KB) | s_idx[64] | s_rel[64][3]      total 148480 B (<320KB WGP)
__global__ __launch_bounds__(256) void k_edge_fused(
    const float* __restrict__ pos, const int* __restrict__ idx, const float* __restrict__ vsd,
    const unsigned short* __restrict__ pw2T,   // [128][512] bf16
    const unsigned short* __restrict__ aw1T,   // [512][128] bf16
    const unsigned short* __restrict__ aw2T,   // [128][512] bf16
    const float* __restrict__ pw1, const float* __restrict__ pb1,
    const float* __restrict__ pg1, const float* __restrict__ pbt1,
    const float* __restrict__ pb2, const float* __restrict__ pg2, const float* __restrict__ pbt2,
    const float* __restrict__ ab1, const float* __restrict__ ag1, const float* __restrict__ abt1,
    const float* __restrict__ ab2, const float* __restrict__ ag2, const float* __restrict__ abt2,
    float* __restrict__ out) {
  extern __shared__ unsigned char smem[];
  unsigned short* s_act   = (unsigned short*)(smem);             // [64][512]
  float*          s_delta = (float*)(smem + 65536);              // [64][128]
  float*          s_alpha = (float*)(smem + 98304);              // [64][128]
  unsigned short* s_ain   = (unsigned short*)(smem + 131072);    // [64][128]
  int*            s_idx   = (int*)(smem + 147456);               // [64]
  float*          s_rel   = (float*)(smem + 147712);             // [64][3]

  int i0 = blockIdx.x * GPB;
  int cnt = NPTS - i0; if (cnt > GPB) cnt = GPB;
  int nrows = cnt * NNB;                      // 63 (or 21 for the tail block)
  int t = threadIdx.x;
  int lane = t & 31, wid = t >> 5, half = lane >> 4, lm = lane & 15;

  // ---- gather neighbor indices + relative positions
  for (int rr = t; rr < nrows; rr += 256) {
    int g = rr / NNB, k = rr - g * NNB;
    int i = i0 + g;
    int j = idx[i * NNB + k];
    s_idx[rr] = j;
    s_rel[rr * 3 + 0] = pos[i * 3 + 0] - pos[j * 3 + 0];
    s_rel[rr * 3 + 1] = pos[i * 3 + 1] - pos[j * 3 + 1];
    s_rel[rr * 3 + 2] = pos[i * 3 + 2] - pos[j * 3 + 2];
  }
  // ---- zero pad rows so GEMM A inputs are clean
  for (int e = nrows * HDIM + t; e < MROWS * HDIM; e += 256) s_act[e] = 0;
  for (int e = nrows * CIN + t; e < MROWS * CIN; e += 256) s_ain[e] = 0;
  __syncthreads();

  // ---- pos_nn stage 1 (K=3, VALU): h = rel @ pw1 + pb1
  for (int e = t; e < nrows * HDIM; e += 256) {
    int r = e >> 9, c = e & 511;
    float h = s_rel[r * 3 + 0] * pw1[c] + s_rel[r * 3 + 1] * pw1[512 + c] +
              s_rel[r * 3 + 2] * pw1[1024 + c] + pb1[c];
    s_act[r * HDIM + c] = f2bf(h);
  }
  __syncthreads();
  ln_elu_512_bf(s_act, pg1, pbt1, lane, wid, nrows);
  __syncthreads();

  // ---- pos_nn stage 2 (WMMA): delta_pre[64,128] = act[64,512] @ pw2 + pb2
  {
    int nt = wid;  // 8 waves <-> 8 N-tiles; one B fragment feeds 4 M-tiles
    const unsigned short* br = pw2T + (nt * 16 + lm) * HDIM;
    v8f zero = {0.f, 0.f, 0.f, 0.f, 0.f, 0.f, 0.f, 0.f};
    v8f acc[4];
#pragma unroll
    for (int mt = 0; mt < 4; ++mt) acc[mt] = zero;
#pragma unroll 4
    for (int k = 0; k < HDIM; k += 32) {
      __builtin_prefetch(br + k + 128, 0, 1);
      v16bf b = fragB(br + k, half);
#pragma unroll
      for (int mt = 0; mt < 4; ++mt) {
        v16bf a = fragA(s_act + (mt * 16 + lm) * HDIM + k, half);
        acc[mt] = __builtin_amdgcn_wmma_f32_16x16x32_bf16(false, a, false, b, (short)0, acc[mt],
                                                          false, false);
      }
    }
    int n = nt * 16 + lm; float bias = pb2[n];
#pragma unroll
    for (int mt = 0; mt < 4; ++mt)
#pragma unroll
      for (int r = 0; r < 8; ++r)
        s_delta[(mt * 16 + half * 8 + r) * CIN + n] = acc[mt][r] + bias;
  }
  __syncthreads();
  ln_elu_128_f32(s_delta, pg2, pbt2, lane, wid, nrows);
  __syncthreads();

  // ---- alpha_in = a_dst[i] - a_src[j] + delta   (bf16 for WMMA A)
  for (int e = t; e < nrows * CIN; e += 256) {
    int r = e >> 7, c = e & 127;
    int i = i0 + r / NNB;
    int j = s_idx[r];
    float v = vsd[i * 384 + 256 + c] - vsd[j * 384 + 128 + c] + s_delta[r * CIN + c];
    s_ain[r * CIN + c] = f2bf(v);
  }
  __syncthreads();

  // ---- attn_nn stage 1 (WMMA): h2_pre[64,512] = ain[64,128] @ aw1 + ab1 -> s_act
  for (int it = 0; it < 4; ++it) {
    int nt = wid * 4 + it;  // 32 N-tiles over 512 cols
    const unsigned short* br = aw1T + (nt * 16 + lm) * CIN;
    __builtin_prefetch(br, 0, 1);
    v8f zero = {0.f, 0.f, 0.f, 0.f, 0.f, 0.f, 0.f, 0.f};
    v8f acc[4];
#pragma unroll
    for (int mt = 0; mt < 4; ++mt) acc[mt] = zero;
#pragma unroll
    for (int k = 0; k < CIN; k += 32) {
      v16bf b = fragB(br + k, half);
#pragma unroll
      for (int mt = 0; mt < 4; ++mt) {
        v16bf a = fragA(s_ain + (mt * 16 + lm) * CIN + k, half);
        acc[mt] = __builtin_amdgcn_wmma_f32_16x16x32_bf16(false, a, false, b, (short)0, acc[mt],
                                                          false, false);
      }
    }
    int n = nt * 16 + lm; float bias = ab1[n];
#pragma unroll
    for (int mt = 0; mt < 4; ++mt)
#pragma unroll
      for (int r = 0; r < 8; ++r)
        s_act[(mt * 16 + half * 8 + r) * HDIM + n] = f2bf(acc[mt][r] + bias);
  }
  __syncthreads();
  ln_elu_512_bf(s_act, ag1, abt1, lane, wid, nrows);
  __syncthreads();

  // ---- attn_nn stage 2 (WMMA): alpha_pre[64,128] = act[64,512] @ aw2 + ab2 -> s_alpha
  {
    int nt = wid;
    const unsigned short* br = aw2T + (nt * 16 + lm) * HDIM;
    v8f zero = {0.f, 0.f, 0.f, 0.f, 0.f, 0.f, 0.f, 0.f};
    v8f acc[4];
#pragma unroll
    for (int mt = 0; mt < 4; ++mt) acc[mt] = zero;
#pragma unroll 4
    for (int k = 0; k < HDIM; k += 32) {
      __builtin_prefetch(br + k + 128, 0, 1);
      v16bf b = fragB(br + k, half);
#pragma unroll
      for (int mt = 0; mt < 4; ++mt) {
        v16bf a = fragA(s_act + (mt * 16 + lm) * HDIM + k, half);
        acc[mt] = __builtin_amdgcn_wmma_f32_16x16x32_bf16(false, a, false, b, (short)0, acc[mt],
                                                          false, false);
      }
    }
    int n = nt * 16 + lm; float bias = ab2[n];
#pragma unroll
    for (int mt = 0; mt < 4; ++mt)
#pragma unroll
      for (int r = 0; r < 8; ++r)
        s_alpha[(mt * 16 + half * 8 + r) * CIN + n] = acc[mt][r] + bias;
  }
  __syncthreads();
  ln_elu_128_f32(s_alpha, ag2, abt2, lane, wid, nrows);
  __syncthreads();

  // ---- per-channel softmax over 21 neighbors + aggregation, per node in this block
  for (int e = t; e < cnt * CIN; e += 256) {
    int g = e >> 7, c = e & 127;
    int base = g * NNB;
    float m = -1e30f;
    for (int r = 0; r < NNB; ++r) m = fmaxf(m, s_alpha[(base + r) * CIN + c]);
    float ssum = 0.f;
    for (int r = 0; r < NNB; ++r) ssum += __expf(s_alpha[(base + r) * CIN + c] - m);
    float inv = 1.f / ssum;
    float acc = 0.f;
    for (int r = 0; r < NNB; ++r) {
      int rr = base + r;
      int j = s_idx[rr];
      acc += __expf(s_alpha[rr * CIN + c] - m) * inv * (vsd[j * 384 + c] + s_delta[rr * CIN + c]);
    }
    out[(i0 + g) * CIN + c] = acc;
  }
}

// ---------------------------------------------------------------- launch
extern "C" void kernel_launch(void* const* d_in, const int* in_sizes, int n_in,
                              void* d_out, int out_size, void* d_ws, size_t ws_size,
                              hipStream_t stream) {
  const float* x     = (const float*)d_in[0];
  const float* pos   = (const float*)d_in[1];
  const float* W_lin = (const float*)d_in[3];
  const float* W_src = (const float*)d_in[4];
  const float* W_dst = (const float*)d_in[5];
  const float* pw1  = (const float*)d_in[6];
  const float* pb1  = (const float*)d_in[7];
  const float* pg1  = (const float*)d_in[8];
  const float* pbt1 = (const float*)d_in[9];
  const float* pw2  = (const float*)d_in[10];
  const float* pb2  = (const float*)d_in[11];
  const float* pg2  = (const float*)d_in[12];
  const float* pbt2 = (const float*)d_in[13];
  const float* aw1  = (const float*)d_in[14];
  const float* ab1  = (const float*)d_in[15];
  const float* ag1  = (const float*)d_in[16];
  const float* abt1 = (const float*)d_in[17];
  const float* aw2  = (const float*)d_in[18];
  const float* ab2  = (const float*)d_in[19];
  const float* ag2  = (const float*)d_in[20];
  const float* abt2 = (const float*)d_in[21];
  float* out = (float*)d_out;

  unsigned char* ws = (unsigned char*)d_ws;
  size_t off = 0;
  auto carve = [&](size_t bytes) -> void* {
    void* p = ws + off;
    off = (off + bytes + 255) & ~(size_t)255;
    return p;
  };
  int*            idx   = (int*)carve((size_t)NPTS * NNB * 4);
  unsigned short* xb    = (unsigned short*)carve((size_t)NPTS * CIN * 2);
  unsigned short* wcatT = (unsigned short*)carve((size_t)384 * 128 * 2);
  unsigned short* pw2T  = (unsigned short*)carve((size_t)128 * 512 * 2);
  unsigned short* aw1T  = (unsigned short*)carve((size_t)512 * 128 * 2);
  unsigned short* aw2T  = (unsigned short*)carve((size_t)128 * 512 * 2);
  float*          vsd   = (float*)carve((size_t)NPTS * 384 * 4);

  k_cast_bf16<<<(NPTS * CIN + 255) / 256, 256, 0, stream>>>(x, xb, NPTS * CIN);
  k_wcat<<<(384 * 128 + 255) / 256, 256, 0, stream>>>(W_lin, W_src, W_dst, wcatT);
  k_transpose_bf16<<<(512 * 128 + 255) / 256, 256, 0, stream>>>(pw2, pw2T, 512, 128);
  k_transpose_bf16<<<(128 * 512 + 255) / 256, 256, 0, stream>>>(aw1, aw1T, 128, 512);
  k_transpose_bf16<<<(512 * 128 + 255) / 256, 256, 0, stream>>>(aw2, aw2T, 512, 128);
  k_knn<<<NPTS / 128, 128, 0, stream>>>(pos, idx);
  k_node_gemm<<<768, 256, 0, stream>>>(xb, wcatT, vsd);

  int nblk = (NPTS + GPB - 1) / GPB;                       // 1366 blocks of 3 nodes
  size_t smem = 65536 + 32768 + 32768 + 16384 + 256 + 768; // 148480 B (<320KB WGP limit)
  k_edge_fused<<<nblk, 256, smem, stream>>>(pos, idx, vsd, pw2T, aw1T, aw2T,
      pw1, pb1, pg1, pbt1, pb2, pg2, pbt2, ab1, ag1, abt1, ab2, ag2, abt2, out);
}
